// T5InfiniAttention_28587302323105
// MI455X (gfx1250) — matrix operand order repaired and checked by reading, not compiled, not run on hardware
//
#include <hip/hip_runtime.h>

typedef __bf16 bf16_t;
typedef __attribute__((ext_vector_type(16))) __bf16 v16bf;
typedef __attribute__((ext_vector_type(8)))  float  v8f;
typedef __attribute__((ext_vector_type(4))) unsigned int u32x4;
typedef __attribute__((ext_vector_type(8))) int i32x8;
typedef __attribute__((ext_vector_type(4))) int i32x4;

constexpr int kB = 32, kS = 512, kDM = 1024, kH = 16, kD = 64, kC = 8, kG = 4;
constexpr int kNB = 32, kMD = 128;
constexpr int kHD   = kH * kD;   // 1024
constexpr int kRows = kB * kS;   // 16384

// ---------------- WMMA fragment helpers (wave32, 16x16x32 bf16) ----------------
// A (16x32, 16-bit): lane holds row M=lane&15; VGPR j pair K offsets per ISA table.
__device__ __forceinline__ int a_koff(int j, int hi) {
  return 2 * j + (j < 4 ? (hi ? 8 : 0) : (hi ? 16 : 8));
}
// B (32x16, 16-bit): lane holds col N=lane&15; VGPR j pair K offsets same as A.
__device__ __forceinline__ int b_koff(int j, int hi) {
  return 2 * j + (j < 4 ? (hi ? 8 : 0) : (hi ? 16 : 8));
}

__device__ __forceinline__ v16bf load_pairs_a(const bf16_t* rowbase, int hi) {
  union { v16bf v; unsigned int u[8]; } r;
#pragma unroll
  for (int j = 0; j < 8; ++j)
    r.u[j] = *(const unsigned int*)(rowbase + a_koff(j, hi));
  return r.v;
}
__device__ __forceinline__ v16bf load_pairs_b(const bf16_t* rowbase, int hi) {
  union { v16bf v; unsigned int u[8]; } r;
#pragma unroll
  for (int j = 0; j < 8; ++j)
    r.u[j] = *(const unsigned int*)(rowbase + b_koff(j, hi));
  return r.v;
}
// B fragment from a row-major [K][rowstride] LDS tile: element (K,N) = base[K*stride+N]
__device__ __forceinline__ v16bf load_pairs_b_strided(const bf16_t* base, int n,
                                                      int rowstride, int hi) {
  union { v16bf v; unsigned short s[16]; } r;
#pragma unroll
  for (int j = 0; j < 8; ++j) {
    int k = b_koff(j, hi);
    r.s[2 * j]     = *(const unsigned short*)(base + (size_t)k * rowstride + n);
    r.s[2 * j + 1] = *(const unsigned short*)(base + (size_t)(k + 1) * rowstride + n);
  }
  return r.v;
}
__device__ __forceinline__ v8f wmma_bf16(v16bf a, v16bf b, v8f c) {
  return __builtin_amdgcn_wmma_f32_16x16x32_bf16(false, a, false, b, (short)0, c,
                                                 false, false);
}

// ---------------- cross-lane butterflies via ds_swizzle (no index VGPR) ----------------
template <int IMM> __device__ __forceinline__ float swz(float x) {
  return __int_as_float(__builtin_amdgcn_ds_swizzle(__float_as_int(x), IMM));
}
__device__ __forceinline__ float redmax16(float x) {
  x = fmaxf(x, swz<0x041F>(x));   // xor 1
  x = fmaxf(x, swz<0x081F>(x));   // xor 2
  x = fmaxf(x, swz<0x101F>(x));   // xor 4
  x = fmaxf(x, swz<0x201F>(x));   // xor 8
  return x;
}
__device__ __forceinline__ float redsum16(float x) {
  x += swz<0x041F>(x);
  x += swz<0x081F>(x);
  x += swz<0x101F>(x);
  x += swz<0x201F>(x);
  return x;
}

// ---------------- Tensor Data Mover: 2-D tile load Global -> LDS ----------------
__device__ __forceinline__ unsigned lds_byte_off(const void* p) {
  return (unsigned)(unsigned long long)(size_t)p;   // low 32 bits = LDS byte offset
}
// tile_d0 contiguous elements per line, tile_d1 lines, line stride in elements (2B data)
__device__ __forceinline__ void tdm_load_2d(unsigned lds_off, const void* g,
                                            unsigned tile_d0, unsigned tile_d1,
                                            unsigned stride_elems) {
  unsigned long long ga = (unsigned long long)(size_t)g;
  unsigned td0 = stride_elems;    // tensor_dim0 >= tile_d0 (no OOB clamp)
  unsigned td1 = 1u << 16;        // tensor_dim1 (generous)
  u32x4 g0;
  g0[0] = 1u;                                    // count=1 user descriptor
  g0[1] = lds_off;                               // lds_addr (bytes)
  g0[2] = (unsigned)ga;                          // global_addr[31:0]
  g0[3] = ((unsigned)(ga >> 32) & 0x01FFFFFFu)   // global_addr[56:32]
        | (2u << 30);                            // type = 2 ("image")
  i32x8 g1;
  g1[0] = (int)(1u << 16);                       // data_size = 1 -> 2 bytes
  g1[1] = (int)((td0 & 0xFFFFu) << 16);          // tensor_dim0[15:0] @ bits 63:48
  g1[2] = (int)(((td0 >> 16) & 0xFFFFu) | ((td1 & 0xFFFFu) << 16));
  g1[3] = (int)(((td1 >> 16) & 0xFFFFu) | ((tile_d0 & 0xFFFFu) << 16)); // tile_dim0
  g1[4] = (int)(tile_d1 & 0xFFFFu);              // tile_dim1 (tile_dim2 = 0)
  g1[5] = (int)stride_elems;                     // tensor_dim0_stride[31:0]
  g1[6] = 0;
  g1[7] = 0;
  i32x4 gz4 = {0, 0, 0, 0};
  i32x8 gz8 = {0, 0, 0, 0, 0, 0, 0, 0};
  // 6-arg form on this toolchain: (g0, g1, g2, g3, g4, cpol)
  __builtin_amdgcn_tensor_load_to_lds(g0, g1, gz4, gz4, gz8, 0);
}

// ---------------- fp32 -> bf16 conversion ----------------
__global__ __launch_bounds__(256) void cvt_f32_bf16(const float* __restrict__ s,
                                                    bf16_t* __restrict__ d, int n) {
  int i = (blockIdx.x * 256 + threadIdx.x) * 4;
  if (i + 3 < n) {
    float4 f = *(const float4*)(s + i);
    union { bf16_t b[4]; uint2 u; } t;
    t.b[0] = (bf16_t)f.x; t.b[1] = (bf16_t)f.y;
    t.b[2] = (bf16_t)f.z; t.b[3] = (bf16_t)f.w;
    *(uint2*)(d + i) = t.u;
  } else {
    for (int k = i; k < n; ++k) d[k] = (bf16_t)s[k];
  }
}

// ---------------- T5 relative-position bucket LUT: delta in [-511,511] ----------------
__global__ void build_lut(int* __restrict__ lut) {
  int i = blockIdx.x * blockDim.x + threadIdx.x;
  if (i >= 2 * kS - 1) return;
  int rp  = i - (kS - 1);            // key - query
  int nb  = kNB / 2;                 // 16
  int bkt = (rp > 0) ? nb : 0;
  int rpa = rp < 0 ? -rp : rp;
  int mex = nb / 2;                  // 8
  int val;
  if (rpa < mex) {
    val = rpa;
  } else {
    float l = __logf((float)rpa / (float)mex) / __logf((float)kMD / (float)mex);
    int large = mex + (int)(l * (float)(nb - mex));
    val = large < nb - 1 ? large : nb - 1;
  }
  lut[i] = bkt + val;
}

// ---------------- 128x128-tile bf16 WMMA GEMM, TDM double-buffered ----------------
template <bool OUT_F32>
__global__ __launch_bounds__(256)
void gemm_bf16_128x128(const bf16_t* __restrict__ A, const bf16_t* __restrict__ W,
                       void* __restrict__ Cout, int M, int N, int K) {
  __shared__ __align__(16) bf16_t sA[2][128][32];   // [m][k]
  __shared__ __align__(16) bf16_t sB[2][32][128];   // [k][n] (row-major, TDM direct)
  const int tid = threadIdx.x;
  const int lane = tid & 31, w = tid >> 5;
  const int hi = lane >> 4, ln = lane & 15;
  const int wm = w >> 1, wn = w & 1;                // 4x2 wave grid, wave tile 32x64
  const int mblk = blockIdx.x * 128, nblk = blockIdx.y * 128;

  const v8f vzero = {0.f, 0.f, 0.f, 0.f, 0.f, 0.f, 0.f, 0.f};
  v8f acc[2][4];
#pragma unroll
  for (int mi = 0; mi < 2; ++mi)
#pragma unroll
    for (int ni = 0; ni < 4; ++ni) acc[mi][ni] = vzero;

  const int nk = K / 32;
  if (w == 0) {   // wave 0 drives the Tensor Data Mover
    tdm_load_2d(lds_byte_off(&sA[0][0][0]), A + (size_t)mblk * K, 32, 128, K);
    tdm_load_2d(lds_byte_off(&sB[0][0][0]), W + nblk, 128, 32, N);
  }
  for (int ki = 0; ki < nk; ++ki) {
    const int buf = ki & 1;
    if (w == 0) {
      if (ki + 1 < nk) {
        const int kb2 = (ki + 1) * 32;
        tdm_load_2d(lds_byte_off(&sA[buf ^ 1][0][0]),
                    A + (size_t)mblk * K + kb2, 32, 128, K);
        tdm_load_2d(lds_byte_off(&sB[buf ^ 1][0][0]),
                    W + (size_t)kb2 * N + nblk, 128, 32, N);
        __builtin_amdgcn_s_wait_tensorcnt(2);   // retire current tile's pair
      } else {
        __builtin_amdgcn_s_wait_tensorcnt(0);
      }
    }
    __syncthreads();

    v16bf af[2], bfr[4];
#pragma unroll
    for (int mi = 0; mi < 2; ++mi)
      af[mi] = load_pairs_a(&sA[buf][wm * 32 + mi * 16 + ln][0], hi);
#pragma unroll
    for (int ni = 0; ni < 4; ++ni)
      bfr[ni] = load_pairs_b_strided(&sB[buf][0][0], wn * 64 + ni * 16 + ln, 128, hi);
#pragma unroll
    for (int mi = 0; mi < 2; ++mi)
#pragma unroll
      for (int ni = 0; ni < 4; ++ni)
        acc[mi][ni] = wmma_bf16(af[mi], bfr[ni], acc[mi][ni]);
    __syncthreads();
  }

#pragma unroll
  for (int mi = 0; mi < 2; ++mi)
#pragma unroll
    for (int ni = 0; ni < 4; ++ni)
#pragma unroll
      for (int r = 0; r < 8; ++r) {
        int row = mblk + wm * 32 + mi * 16 + r + hi * 8;
        int col = nblk + wn * 64 + ni * 16 + ln;
        float v = acc[mi][ni][r];
        if (OUT_F32) ((float*)Cout)[(size_t)row * N + col] = v;
        else         ((bf16_t*)Cout)[(size_t)row * N + col] = (bf16_t)v;
      }
}

// ---------------- per-(g,h,c) partial M = sk^T v and z = sum(sk) ----------------
__global__ __launch_bounds__(256)
void mz_partial(const bf16_t* __restrict__ Kb, const bf16_t* __restrict__ Vb,
                float* __restrict__ Mpart, float* __restrict__ zpart) {
  __shared__ float skt[64][65];
  __shared__ float vtt[64][65];
  const int bi = blockIdx.x;                   // (g*H + h)*C + c
  const int c = bi & 7, h = (bi >> 3) & 15, g = bi >> 7;
  const int b = g * kC + c;
  const int tid = threadIdx.x;
  const int ti = tid >> 4, tj = tid & 15;

  float acc[4][4];
#pragma unroll
  for (int i = 0; i < 4; ++i)
#pragma unroll
    for (int j = 0; j < 4; ++j) acc[i][j] = 0.f;
  float zacc = 0.f;

  for (int s0 = 0; s0 < kS; s0 += 64) {
    __syncthreads();
    {
      int s = tid >> 2, dch = (tid & 3) * 16;
      const bf16_t* kg = Kb + (size_t)(b * kS + s0 + s) * kHD + h * kD + dch;
      const bf16_t* vg = Vb + (size_t)(b * kS + s0 + s) * kHD + h * kD + dch;
      union { uint4 q[2]; bf16_t bv[16]; } tk, tv;
      tk.q[0] = *(const uint4*)kg; tk.q[1] = *(const uint4*)(kg + 8);
      tv.q[0] = *(const uint4*)vg; tv.q[1] = *(const uint4*)(vg + 8);
      if (s0 + 64 < kS) {
        __builtin_prefetch(kg + (size_t)64 * kHD, 0, 0);
        __builtin_prefetch(vg + (size_t)64 * kHD, 0, 0);
      }
#pragma unroll
      for (int e = 0; e < 16; ++e) {
        float x = (float)tk.bv[e];
        skt[s][dch + e] = x > 0.f ? x + 1.f : __expf(x);   // elu(k)+1
        vtt[s][dch + e] = (float)tv.bv[e];
      }
    }
    __syncthreads();

#pragma unroll 2
    for (int s = 0; s < 64; ++s) {
      float av[4], bv[4];
#pragma unroll
      for (int i = 0; i < 4; ++i) av[i] = skt[s][ti * 4 + i];
#pragma unroll
      for (int j = 0; j < 4; ++j) bv[j] = vtt[s][tj * 4 + j];
#pragma unroll
      for (int i = 0; i < 4; ++i)
#pragma unroll
        for (int j = 0; j < 4; ++j) acc[i][j] = fmaf(av[i], bv[j], acc[i][j]);
    }
    if (tid < 64) {
      float zz = 0.f;
      for (int s = 0; s < 64; ++s) zz += skt[s][tid];
      zacc += zz;
    }
  }

  float* mp = Mpart + (size_t)bi * (kD * kD);
#pragma unroll
  for (int i = 0; i < 4; ++i)
#pragma unroll
    for (int j = 0; j < 4; ++j)
      mp[(ti * 4 + i) * kD + tj * 4 + j] = acc[i][j];
  if (tid < 64) zpart[bi * kD + tid] = zacc;
}

__global__ __launch_bounds__(256)
void mz_reduce(const float* __restrict__ Mpart, const float* __restrict__ zpart,
               float* __restrict__ Mfull, float* __restrict__ zfull) {
  int i = blockIdx.x * 256 + threadIdx.x;
  const int nM = kG * kH * kD * kD;
  if (i < nM) {
    int gh = i >> 12, idx = i & 4095;
    float s = 0.f;
#pragma unroll
    for (int c = 0; c < kC; ++c) s += Mpart[(size_t)(gh * kC + c) * 4096 + idx];
    Mfull[i] = s;
  } else {
    int j = i - nM;
    if (j < kG * kH * kD) {
      int gh = j >> 6, d = j & 63;
      float s = 0.f;
#pragma unroll
      for (int c = 0; c < kC; ++c) s += zpart[(gh * kC + c) * kD + d];
      zfull[j] = s;
    }
  }
}

// ---------------- flash attention + compressive memory + gate (TDM K/V tiles) ----------------
__global__ __launch_bounds__(128)
void attn_kernel(const bf16_t* __restrict__ Qb, const bf16_t* __restrict__ Kb,
                 const bf16_t* __restrict__ Vb, const float* __restrict__ Mfull,
                 const float* __restrict__ zfull, const int* __restrict__ lut,
                 const float* __restrict__ rel_emb, const float* __restrict__ beta,
                 bf16_t* __restrict__ AO) {
  __shared__ __align__(16) bf16_t kt[2][32][64];   // [key][d], double-buffered
  __shared__ __align__(16) bf16_t vkd[2][32][64];  // [key][d], double-buffered
  __shared__ __align__(16) bf16_t ps[4][16][32];   // per-wave P staging
  __shared__ __align__(16) bf16_t mT[64][64];      // M transposed: [e][d]
  __shared__ float  zs[kD];
  __shared__ float  bias_h[kNB];
  __shared__ int    lut_s[2 * kS - 1];
  __shared__ float  den_s[4][16];

  const int hb = blockIdx.x;                 // ((g*C + c)*H + h)
  const int h = hb & 15, c = (hb >> 4) & 7, g = hb >> 7;
  const int b = g * kC + c;
  const int gh = g * kH + h;
  const int qrow0 = blockIdx.y * 64;

  const int tid = threadIdx.x;
  const int lane = tid & 31, w = tid >> 5;
  const int hi = lane >> 4, ln = lane & 15;

  const bf16_t* Kbase = Kb + (size_t)b * kS * kHD + h * kD;
  const bf16_t* Vbase = Vb + (size_t)b * kS * kHD + h * kD;

  if (w == 0) {   // prime first K/V tiles via TDM while we stage everything else
    tdm_load_2d(lds_byte_off(&kt[0][0][0]),  Kbase, 64, 32, kHD);
    tdm_load_2d(lds_byte_off(&vkd[0][0][0]), Vbase, 64, 32, kHD);
  }

  for (int i = tid; i < 2 * kS - 1; i += 128) lut_s[i] = lut[i];
  if (tid < kNB) bias_h[tid] = rel_emb[tid * kH + h];
  if (tid < kD) zs[tid] = zfull[gh * kD + tid];
  for (int i = tid; i < kD * kD; i += 128) {
    int e = i >> 6, d = i & 63;
    mT[e][d] = (bf16_t)Mfull[((size_t)gh * kD + d) * kD + e];
  }
  __syncthreads();

  const float gate = 1.f / (1.f + __expf(-beta[h]));

  // Q fragments (A layout) + sq = elu(q)+1 fragments + denominator partial
  const size_t growQ = (size_t)(b * kS + qrow0 + w * 16 + ln) * kHD + h * kD;
  v16bf qa[2], sqa[2];
  float denp = 0.f;
  {
#pragma unroll
    for (int kc = 0; kc < 2; ++kc) {
      union { v16bf v; unsigned int u[8]; bf16_t bb[16]; } uq, us;
#pragma unroll
      for (int j = 0; j < 8; ++j) {
        int k = kc * 32 + a_koff(j, hi);
        union { unsigned int u; bf16_t b2[2]; } t;
        t.u = *(const unsigned int*)(Qb + growQ + k);
        uq.u[j] = t.u;
#pragma unroll
        for (int p2 = 0; p2 < 2; ++p2) {
          float x = (float)t.b2[p2];
          float sv = x > 0.f ? x + 1.f : __expf(x);
          us.bb[2 * j + p2] = (bf16_t)sv;
          denp += sv * zs[k + p2];
        }
      }
      qa[kc] = uq.v; sqa[kc] = us.v;
    }
  }
  denp += swz<0x401F>(denp);                 // xor 16: fold the two half-waves
  if (lane < 16) den_s[w][lane] = denp;

  float rmax_[8], rsum_[8];
#pragma unroll
  for (int r = 0; r < 8; ++r) { rmax_[r] = -1e30f; rsum_[r] = 0.f; }
  const v8f vzero = {0.f, 0.f, 0.f, 0.f, 0.f, 0.f, 0.f, 0.f};
  v8f oacc[4] = {vzero, vzero, vzero, vzero};

  for (int it = 0; it < kS / 32; ++it) {
    const int kt0 = it * 32;
    const int buf = it & 1;
    if (w == 0) {
      if (it + 1 < kS / 32) {
        const bf16_t* kg = Kbase + (size_t)(kt0 + 32) * kHD;
        const bf16_t* vg = Vbase + (size_t)(kt0 + 32) * kHD;
        tdm_load_2d(lds_byte_off(&kt[buf ^ 1][0][0]),  kg, 64, 32, kHD);
        tdm_load_2d(lds_byte_off(&vkd[buf ^ 1][0][0]), vg, 64, 32, kHD);
        __builtin_amdgcn_s_wait_tensorcnt(2);   // current tile's pair done
      } else {
        __builtin_amdgcn_s_wait_tensorcnt(0);
      }
    }
    __syncthreads();

    // S = bias + Q K^T (two 16x16 N-fragments)
    v8f sc[2];
#pragma unroll
    for (int nf = 0; nf < 2; ++nf) {
      v8f ci;
#pragma unroll
      for (int r = 0; r < 8; ++r) {
        int m = r + hi * 8;
        int qr = qrow0 + w * 16 + m;
        int ky = kt0 + nf * 16 + ln;
        ci[r] = bias_h[lut_s[ky - qr + (kS - 1)]];
      }
#pragma unroll
      for (int kc = 0; kc < 2; ++kc) {
        v16bf kfrag = load_pairs_b(&kt[buf][nf * 16 + ln][kc * 32], hi);
        ci = wmma_bf16(qa[kc], kfrag, ci);
      }
      sc[nf] = ci;
    }

    // online softmax (row reductions across the 16 lanes of each half-wave)
    float corr_[8];
#pragma unroll
    for (int r = 0; r < 8; ++r) {
      float mx = redmax16(fmaxf(sc[0][r], sc[1][r]));
      float nmax = fmaxf(rmax_[r], mx);
      float p0 = __expf(sc[0][r] - nmax);
      float p1 = __expf(sc[1][r] - nmax);
      sc[0][r] = p0; sc[1][r] = p1;
      float sm = redsum16(p0 + p1);
      float corr = __expf(rmax_[r] - nmax);
      rsum_[r] = rsum_[r] * corr + sm;
      rmax_[r] = nmax;
      corr_[r] = corr;
    }
#pragma unroll
    for (int nf = 0; nf < 4; ++nf)
#pragma unroll
      for (int r = 0; r < 8; ++r) oacc[nf][r] *= corr_[r];

    // stage P (C layout -> A layout via per-wave LDS)
#pragma unroll
    for (int r = 0; r < 8; ++r) {
      int m = r + hi * 8;
      ps[w][m][ln]      = (bf16_t)sc[0][r];
      ps[w][m][16 + ln] = (bf16_t)sc[1][r];
    }
    v16bf pa = load_pairs_a(&ps[w][ln][0], hi);
#pragma unroll
    for (int nf = 0; nf < 4; ++nf) {
      v16bf vfrag = load_pairs_b_strided(&vkd[buf][0][0], nf * 16 + ln, 64, hi);
      oacc[nf] = wmma_bf16(pa, vfrag, oacc[nf]);
    }
    __syncthreads();   // everyone done with buf before TDM overwrites it
  }

  // compressive memory numerator: sq @ M
  v8f nm[4];
#pragma unroll
  for (int nf = 0; nf < 4; ++nf) {
    v8f cc = vzero;
#pragma unroll
    for (int kc = 0; kc < 2; ++kc) {
      v16bf mfrag = load_pairs_b(&mT[nf * 16 + ln][kc * 32], hi);
      cc = wmma_bf16(sqa[kc], mfrag, cc);
    }
    nm[nf] = cc;
  }

  // gate-combine and store (row = b*S+s, col = h*64+d)
#pragma unroll
  for (int nf = 0; nf < 4; ++nf)
#pragma unroll
    for (int r = 0; r < 8; ++r) {
      int m = r + hi * 8;
      int qr = qrow0 + w * 16 + m;
      float adot = oacc[nf][r] / rsum_[r];
      float amem = nm[nf][r] / (den_s[w][m] + 1e-6f);
      float o = gate * amem + (1.f - gate) * adot;
      AO[(size_t)(b * kS + qr) * kHD + h * kD + nf * 16 + ln] = (bf16_t)o;
    }
}

// ---------------- host launch ----------------
extern "C" void kernel_launch(void* const* d_in, const int* in_sizes, int n_in,
                              void* d_out, int out_size, void* d_ws, size_t ws_size,
                              hipStream_t stream) {
  (void)in_sizes; (void)n_in; (void)out_size; (void)ws_size;
  const float* hidden  = (const float*)d_in[0];
  const float* Wq      = (const float*)d_in[1];
  const float* Wk      = (const float*)d_in[2];
  const float* Wv      = (const float*)d_in[3];
  const float* Wo      = (const float*)d_in[4];
  const float* rel_emb = (const float*)d_in[5];
  const float* beta    = (const float*)d_in[6];
  float* out = (float*)d_out;

  char* ws = (char*)d_ws;
  size_t off = 0;
  auto alloc = [&](size_t bytes) -> void* {
    void* p = ws + off;
    off += (bytes + 255) & ~(size_t)255;
    return p;
  };
  bf16_t* Xb    = (bf16_t*)alloc((size_t)kRows * kDM * 2);  // reused as AO later
  bf16_t* Wqb   = (bf16_t*)alloc((size_t)kDM * kHD * 2);
  bf16_t* Wkb   = (bf16_t*)alloc((size_t)kDM * kHD * 2);
  bf16_t* Wvb   = (bf16_t*)alloc((size_t)kDM * kHD * 2);
  bf16_t* Wob   = (bf16_t*)alloc((size_t)kHD * kDM * 2);
  bf16_t* Qb    = (bf16_t*)alloc((size_t)kRows * kHD * 2);
  bf16_t* Kb    = (bf16_t*)alloc((size_t)kRows * kHD * 2);
  bf16_t* Vb    = (bf16_t*)alloc((size_t)kRows * kHD * 2);
  float*  Mpart = (float*)alloc((size_t)kG * kH * kC * kD * kD * 4);
  float*  zpart = (float*)alloc((size_t)kG * kH * kC * kD * 4);
  float*  Mfull = (float*)alloc((size_t)kG * kH * kD * kD * 4);
  float*  zfull = (float*)alloc((size_t)kG * kH * kD * 4);
  int*    lut   = (int*)alloc((2 * kS - 1) * 4);

  const int nX = kRows * kDM, nW = kDM * kHD;
  cvt_f32_bf16<<<(nX / 4 + 255) / 256, 256, 0, stream>>>(hidden, Xb, nX);
  cvt_f32_bf16<<<(nW / 4 + 255) / 256, 256, 0, stream>>>(Wq, Wqb, nW);
  cvt_f32_bf16<<<(nW / 4 + 255) / 256, 256, 0, stream>>>(Wk, Wkb, nW);
  cvt_f32_bf16<<<(nW / 4 + 255) / 256, 256, 0, stream>>>(Wv, Wvb, nW);
  cvt_f32_bf16<<<(nW / 4 + 255) / 256, 256, 0, stream>>>(Wo, Wob, nW);
  build_lut<<<4, 256, 0, stream>>>(lut);

  dim3 ggrid(kRows / 128, kHD / 128);
  gemm_bf16_128x128<false><<<ggrid, 256, 0, stream>>>(Xb, Wqb, Qb, kRows, kHD, kDM);
  gemm_bf16_128x128<false><<<ggrid, 256, 0, stream>>>(Xb, Wkb, Kb, kRows, kHD, kDM);
  gemm_bf16_128x128<false><<<ggrid, 256, 0, stream>>>(Xb, Wvb, Vb, kRows, kHD, kDM);

  mz_partial<<<kG * kH * kC, 256, 0, stream>>>(Kb, Vb, Mpart, zpart);
  const int nr = kG * kH * kD * kD + kG * kH * kD;
  mz_reduce<<<(nr + 255) / 256, 256, 0, stream>>>(Mpart, zpart, Mfull, zfull);

  bf16_t* AO = Xb;  // X no longer needed; alias (stream-ordered)
  attn_kernel<<<dim3(kG * kC * kH, kS / 64), 128, 0, stream>>>(
      Qb, Kb, Vb, Mfull, zfull, lut, rel_emb, beta, AO);

  gemm_bf16_128x128<true><<<ggrid, 256, 0, stream>>>(AO, Wob, out, kRows, kDM, kHD);
}
